// EfficientAttention_60327110640431
// MI455X (gfx1250) — matrix operand order repaired and checked by reading, not compiled, not run on hardware
//
#include <hip/hip_runtime.h>
#include <hip/hip_bf16.h>

// ---------------- types ----------------
typedef __attribute__((ext_vector_type(16))) __bf16 bf16x16;
typedef __attribute__((ext_vector_type(8)))  __bf16 bf16x8;
typedef __attribute__((ext_vector_type(4)))  __bf16 bf16x4;
typedef __attribute__((ext_vector_type(8)))  float  f32x8;

union AFrag { bf16x16 v; bf16x8 h[2]; };

__device__ __forceinline__ f32x8 wmma_bf16(const AFrag& a, const AFrag& b, f32x8 c) {
    return __builtin_amdgcn_wmma_f32_16x16x32_bf16(
        /*neg_a=*/false, a.v, /*neg_b=*/false, b.v,
        /*c_mod=*/(short)0, c, /*reuse_a=*/false, /*reuse_b=*/false);
}

// LDS byte-offset of a pointer into __shared__ memory (addrspace(3) is 32-bit).
typedef const void __attribute__((address_space(3)))* lds_cptr;
__device__ __forceinline__ unsigned lds_off_u32(const void* p) {
    return (unsigned)(unsigned long long)(lds_cptr)p;
}

// ---------------- elementwise cast ----------------
__global__ void cast_f32_bf16(const float* __restrict__ src, __bf16* __restrict__ dst, long n) {
    long i = (long)blockIdx.x * blockDim.x + threadIdx.x;
    long stride = (long)gridDim.x * blockDim.x;
    for (; i < n; i += stride) dst[i] = (__bf16)src[i];
}

// ---------------- generic bf16 WMMA GEMM: C(f32) = A(MxK) * B(KxN) ----------------
// Block tile 128x128x32, 256 threads (8 waves), wave tile 64x32 (8 WMMAs/chunk).
// A tile staged via CDNA5 global_load_async_to_lds_b128 (ASYNCcnt); B tile staged
// transposed through VGPRs so both fragments read as contiguous ds_load_b128.
#define GBM 128
#define GBN 128
#define GBK 32
#define LDSP (GBK + 8)

__global__ __launch_bounds__(256) void gemm_bf16_f32(
    const __bf16* __restrict__ A, const __bf16* __restrict__ B,
    float* __restrict__ C, int M, int N, int K)
{
    __shared__ __bf16 Alds[GBM][LDSP];   // row-major A tile, padded
    __shared__ __bf16 Blds[GBN][LDSP];   // B tile stored transposed [n][k], padded

    const int tid  = threadIdx.x;
    const int lane = tid & 31;
    const int wave = tid >> 5;
    const int m0 = blockIdx.y * GBM;
    const int n0 = blockIdx.x * GBN;
    const int wm0 = (wave >> 2) * 64;    // 2 wave rows of 64
    const int wn0 = (wave & 3) * 32;     // 4 wave cols of 32
    const int half = lane >> 4;
    const int l16  = lane & 15;

    f32x8 acc[4][2] = {};

    for (int kb = 0; kb < K; kb += GBK) {
        // ---- stage A tile (128x32 bf16 = 8KB) via async global->LDS b128 copies ----
        {
            unsigned long long gbase =
                (unsigned long long)(const void*)(A + (size_t)m0 * K + kb);
            for (int s = 0; s < 2; ++s) {
                int idx = s * 256 + tid;           // 512 x 16B chunks
                int row = idx >> 2;
                int kk  = (idx & 3) * 8;
                unsigned goff = (unsigned)((row * K + kk) * 2);
                unsigned loff = lds_off_u32(&Alds[row][kk]);
                asm volatile("global_load_async_to_lds_b128 %0, %1, %2"
                             :: "v"(loff), "v"(goff), "s"(gbase) : "memory");
            }
        }
        // ---- stage B tile transposed: 32x128, 8B vector loads, scalar LDS writes ----
        for (int s = 0; s < 4; ++s) {
            int k  = s * 8 + (tid >> 5);
            int n4 = (tid & 31) * 4;
            bf16x4 t = *(const bf16x4*)&B[(size_t)(kb + k) * N + n0 + n4];
            Blds[n4 + 0][k] = t[0];
            Blds[n4 + 1][k] = t[1];
            Blds[n4 + 2][k] = t[2];
            Blds[n4 + 3][k] = t[3];
        }
        asm volatile("s_wait_asynccnt 0x0" ::: "memory");
        __syncthreads();

        // ---- fragments + 8 WMMAs per wave ----
        AFrag a[4], bfr[2];
        for (int i = 0; i < 4; ++i) {
            int row = wm0 + i * 16 + l16;
            a[i].h[0] = *(const bf16x8*)&Alds[row][half * 16];
            a[i].h[1] = *(const bf16x8*)&Alds[row][half * 16 + 8];
        }
        for (int j = 0; j < 2; ++j) {
            int col = wn0 + j * 16 + l16;
            bfr[j].h[0] = *(const bf16x8*)&Blds[col][half * 16];
            bfr[j].h[1] = *(const bf16x8*)&Blds[col][half * 16 + 8];
        }
        for (int i = 0; i < 4; ++i)
            for (int j = 0; j < 2; ++j)
                acc[i][j] = wmma_bf16(a[i], bfr[j], acc[i][j]);
        __syncthreads();
    }

    // ---- epilogue: per-i/e rows are 16 lanes contiguous in n ----
    float* crow = C + (size_t)(m0 + wm0 + half * 8) * N + n0 + wn0 + l16;
    for (int i = 0; i < 4; ++i)
        for (int j = 0; j < 2; ++j)
            for (int e = 0; e < 8; ++e)
                crow[(size_t)(i * 16 + e) * N + j * 16] = acc[i][j][e];
}

// ---------------- RoPE + cast + relayout: [B*T, NH*128] f32 -> [B, NH, T, 128] bf16 ----------------
__global__ void rope_relayout(const float* __restrict__ src, __bf16* __restrict__ dst,
                              int B, int T, int NH) {
    const long n = (long)B * T * NH * 64;
    long idx = (long)blockIdx.x * blockDim.x + threadIdx.x;
    long stride = (long)gridDim.x * blockDim.x;
    for (; idx < n; idx += stride) {
        int j  = (int)(idx & 63);
        int h  = (int)((idx >> 6) % NH);
        int t  = (int)((idx / (64L * NH)) % T);
        int b  = (int)( idx / (64L * NH * T));
        size_t so = (((size_t)b * T + t) * NH + h) * 128;
        float x0 = src[so + j];
        float x1 = src[so + 64 + j];
        // inv_freq = 10000^(-2j/128) = exp(-j * ln(10000)/64)
        float freq = (float)t * __expf(-(float)j * 0.14391156831212787f);
        float c = cosf(freq), s = sinf(freq);
        float o0 = x0 * c - x1 * s;     // rotate_half: first half gets -x[j+64]
        float o1 = x1 * c + x0 * s;
        size_t dofs = (((size_t)b * NH + h) * T + t) * 128;
        dst[dofs + j]      = (__bf16)o0;
        dst[dofs + 64 + j] = (__bf16)o1;
    }
}

// ---------------- V cast + transpose: [B*T, KV*128] f32 -> [B, KV, 128, T] bf16 ----------------
__global__ void v_transpose(const float* __restrict__ src, __bf16* __restrict__ dst, int B, int T) {
    const long n = (long)B * T * 4 * 128;
    long idx = (long)blockIdx.x * blockDim.x + threadIdx.x;
    long stride = (long)gridDim.x * blockDim.x;
    for (; idx < n; idx += stride) {
        int d  = (int)(idx & 127);
        int kv = (int)((idx >> 7) & 3);
        int t  = (int)((idx / 512L) % T);
        int b  = (int)( idx / (512L * T));
        dst[(((size_t)b * 4 + kv) * 128 + d) * T + t] =
            (__bf16)src[(((size_t)b * T + t) * 4 + kv) * 128 + d];
    }
}

// ---------------- flash attention: Q[B,H,T,128] x K[B,KV,T,128] x Vt[B,KV,128,T] -> O[B,T,H*128] bf16 ----------------
__global__ __launch_bounds__(128) void attn_fwd(
    const __bf16* __restrict__ Q, const __bf16* __restrict__ Kd,
    const __bf16* __restrict__ Vt, __bf16* __restrict__ O, int T)
{
    constexpr int D = 128, H = 16, KV = 4;
    const int h    = blockIdx.y;
    const int b    = blockIdx.z;
    const int kvh  = h / (H / KV);
    const int lane = threadIdx.x & 31;
    const int wave = threadIdx.x >> 5;
    const int qbase = blockIdx.x * 64 + wave * 16;   // 16 query rows per wave
    const int half  = lane >> 4;
    const int l16   = lane & 15;

    const __bf16* Qh = Q  + ((size_t)(b * H  + h  ) * T) * D;
    const __bf16* Kh = Kd + ((size_t)(b * KV + kvh) * T) * D;
    const __bf16* Vh = Vt + ((size_t)(b * KV + kvh) * D) * T;

    __shared__ __bf16 Plds[4][16][40];   // per-wave P tile (16 q x 32 k), padded

    // Q fragments: 4 d-chunks of 32
    AFrag qf[4];
    {
        const __bf16* qp = Qh + (size_t)(qbase + l16) * D;
        for (int kc = 0; kc < 4; ++kc) {
            qf[kc].h[0] = *(const bf16x8*)&qp[kc * 32 + half * 16];
            qf[kc].h[1] = *(const bf16x8*)&qp[kc * 32 + half * 16 + 8];
        }
    }

    f32x8 acc[8] = {};          // 16 x 128 output tile (8 d-tiles)
    float mrow[8], lrow[8];
    for (int i = 0; i < 8; ++i) { mrow[i] = -1e30f; lrow[i] = 0.f; }
    const float scale = 0.08838834764831845f;   // 1/sqrt(128)

    const int q0   = qbase + half * 8;          // element i covers row q0+i
    const int kend = qbase + 16;                // causal: keys < kend

    for (int kb = 0; kb < kend; kb += 32) {
        // ---- scores: S(16x32) = Qtile(16x128) . K^T chunk ----
        f32x8 s0 = {}, s1 = {};
        for (int kc = 0; kc < 4; ++kc) {
            AFrag b0, b1;
            const __bf16* kp0 = Kh + (size_t)(kb + l16) * D + kc * 32 + half * 16;
            const __bf16* kp1 = Kh + (size_t)(kb + 16 + l16) * D + kc * 32 + half * 16;
            b0.h[0] = *(const bf16x8*)kp0;      b0.h[1] = *(const bf16x8*)(kp0 + 8);
            b1.h[0] = *(const bf16x8*)kp1;      b1.h[1] = *(const bf16x8*)(kp1 + 8);
            s0 = wmma_bf16(qf[kc], b0, s0);
            s1 = wmma_bf16(qf[kc], b1, s1);
        }

        // ---- causal mask + scale ----
        const int key0 = kb + l16, key1 = kb + 16 + l16;
        float p0[8], p1[8], newm[8], alpha[8];
        for (int i = 0; i < 8; ++i) {
            float v0 = s0[i] * scale; if (key0 > q0 + i) v0 = -1e30f;
            float v1 = s1[i] * scale; if (key1 > q0 + i) v1 = -1e30f;
            p0[i] = v0; p1[i] = v1;
        }
        // ---- online softmax: row max / sum via 16-lane xor reductions ----
        for (int i = 0; i < 8; ++i) {
            float mx = fmaxf(p0[i], p1[i]);
            for (int off = 1; off < 16; off <<= 1)
                mx = fmaxf(mx, __shfl_xor(mx, off, 32));
            newm[i]  = fmaxf(mrow[i], mx);
            alpha[i] = __expf(mrow[i] - newm[i]);
            mrow[i]  = newm[i];
        }
        for (int i = 0; i < 8; ++i) {
            p0[i] = __expf(p0[i] - newm[i]);
            p1[i] = __expf(p1[i] - newm[i]);
            float sm = p0[i] + p1[i];
            for (int off = 1; off < 16; off <<= 1)
                sm += __shfl_xor(sm, off, 32);
            lrow[i] = lrow[i] * alpha[i] + sm;
        }
        for (int dt = 0; dt < 8; ++dt)
            for (int i = 0; i < 8; ++i)
                acc[dt][i] *= alpha[i];

        // ---- relayout P: C-fragment -> A-fragment through per-wave LDS ----
        for (int i = 0; i < 8; ++i) {
            Plds[wave][half * 8 + i][l16]      = (__bf16)p0[i];
            Plds[wave][half * 8 + i][16 + l16] = (__bf16)p1[i];
        }
        asm volatile("s_wait_dscnt 0" ::: "memory");
        AFrag pf;
        pf.h[0] = *(const bf16x8*)&Plds[wave][l16][half * 16];
        pf.h[1] = *(const bf16x8*)&Plds[wave][l16][half * 16 + 8];

        // ---- O += P(16x32) . V(32x128) ----
        for (int dt = 0; dt < 8; ++dt) {
            AFrag vf;
            const __bf16* vp = Vh + (size_t)(dt * 16 + l16) * T + kb + half * 16;
            vf.h[0] = *(const bf16x8*)vp;  vf.h[1] = *(const bf16x8*)(vp + 8);
            acc[dt] = wmma_bf16(pf, vf, acc[dt]);
        }
    }

    // ---- epilogue: divide by l, store bf16 to [B,T,H*D] ----
    float inv[8];
    for (int i = 0; i < 8; ++i) inv[i] = 1.0f / lrow[i];
    __bf16* op = O + ((size_t)b * T + qbase) * (H * D) + h * D;
    for (int dt = 0; dt < 8; ++dt)
        for (int i = 0; i < 8; ++i)
            op[(size_t)(half * 8 + i) * (H * D) + dt * 16 + l16] =
                (__bf16)(acc[dt][i] * inv[i]);
}

// ---------------- host launcher ----------------
extern "C" void kernel_launch(void* const* d_in, const int* in_sizes, int n_in,
                              void* d_out, int out_size, void* d_ws, size_t ws_size,
                              hipStream_t stream) {
    (void)in_sizes; (void)n_in; (void)out_size; (void)ws_size;
    const float* x  = (const float*)d_in[0];
    const float* Wq = (const float*)d_in[1];
    const float* Wk = (const float*)d_in[2];
    const float* Wv = (const float*)d_in[3];
    const float* Wo = (const float*)d_in[4];
    float* out = (float*)d_out;

    const int B = 2, T = 2048, C = 2048, H = 16, KV = 4;
    const int M = B * T;            // 4096
    const int Ckv = (C / H) * KV;   // 512

    char* ws = (char*)d_ws;
    size_t off = 0;
    auto alloc = [&](size_t bytes) -> void* {
        void* p = ws + off; off += (bytes + 255) & ~(size_t)255; return p;
    };
    __bf16* xb  = (__bf16*)alloc((size_t)M * C * 2);
    __bf16* Wqb = (__bf16*)alloc((size_t)C * C * 2);
    __bf16* Wkb = (__bf16*)alloc((size_t)C * Ckv * 2);
    __bf16* Wvb = (__bf16*)alloc((size_t)C * Ckv * 2);
    __bf16* Wob = (__bf16*)alloc((size_t)C * C * 2);
    float*  qf  = (float*)alloc((size_t)M * C * 4);
    float*  kf  = (float*)alloc((size_t)M * Ckv * 4);
    float*  vf  = (float*)alloc((size_t)M * Ckv * 4);
    __bf16* qrb = (__bf16*)alloc((size_t)M * C * 2);     // [B,H,T,128]
    __bf16* krb = (__bf16*)alloc((size_t)M * Ckv * 2);   // [B,KV,T,128]
    __bf16* vtb = (__bf16*)alloc((size_t)M * Ckv * 2);   // [B,KV,128,T]
    __bf16* ab  = (__bf16*)qf;   // reuse qf region (dead after RoPE) for attn out bf16

    // 1) casts to bf16
    cast_f32_bf16<<<4096, 256, 0, stream>>>(x,  xb,  (long)M * C);
    cast_f32_bf16<<<4096, 256, 0, stream>>>(Wq, Wqb, (long)C * C);
    cast_f32_bf16<<<2048, 256, 0, stream>>>(Wk, Wkb, (long)C * Ckv);
    cast_f32_bf16<<<2048, 256, 0, stream>>>(Wv, Wvb, (long)C * Ckv);
    cast_f32_bf16<<<4096, 256, 0, stream>>>(Wo, Wob, (long)C * C);

    // 2) QKV projections
    gemm_bf16_f32<<<dim3(C   / GBN, M / GBM), 256, 0, stream>>>(xb, Wqb, qf, M, C,   C);
    gemm_bf16_f32<<<dim3(Ckv / GBN, M / GBM), 256, 0, stream>>>(xb, Wkb, kf, M, Ckv, C);
    gemm_bf16_f32<<<dim3(Ckv / GBN, M / GBM), 256, 0, stream>>>(xb, Wvb, vf, M, Ckv, C);

    // 3) RoPE + relayout, V transpose
    rope_relayout<<<4096, 256, 0, stream>>>(qf, qrb, B, T, H);
    rope_relayout<<<1024, 256, 0, stream>>>(kf, krb, B, T, KV);
    v_transpose  <<<2048, 256, 0, stream>>>(vf, vtb, B, T);

    // 4) flash attention (writes bf16 into ab, which aliases dead qf)
    attn_fwd<<<dim3(T / 64, H, B), 128, 0, stream>>>(qrb, krb, vtb, ab, T);

    // 5) output projection -> f32 d_out
    gemm_bf16_f32<<<dim3(C / GBN, M / GBM), 256, 0, stream>>>(ab, Wob, out, M, C, C);
}
